// EdgeExtractionBasic_23261542875747
// MI455X (gfx1250) — compile-verified, matrix operand order checked by reading, not run on hardware
//
#include <hip/hip_runtime.h>
#include <hip/hip_bf16.h>
#include <math.h>

// ---------------------------------------------------------------------------
// Problem constants (from the reference)
// ---------------------------------------------------------------------------
constexpr int kN  = 10000;     // nodes
constexpr int kE  = 160000;    // edges (multiple of 16 -> no tile tails)
constexpr int kD  = 64;        // node feature dim
constexpr int kR  = 8;
constexpr int kA  = 9;
constexpr int kC  = kR + kA;   // 17
constexpr int kHI = 2 * kD + kC; // 145
constexpr int kH  = 128;
constexpr int kNO2 = 81;       // 9*9

// LDS activation tile: 16 rows x 176 halves (160 used, +16 pad); 352B rows
// keep all 16B vector stores aligned.
#define AST 176

typedef __attribute__((ext_vector_type(16))) _Float16 v16h;
typedef __attribute__((ext_vector_type(8)))  _Float16 v8h;
typedef __attribute__((ext_vector_type(8)))  float    v8f;

#define WAIT_DS() asm volatile("s_wait_dscnt 0" ::: "memory")

// ---------------------------------------------------------------------------
// Fragment helpers (CDNA5 WMMA f16 16x16x32 layouts, CDNA5 ISA)
// A (16x32, f16): lane L -> row = L&15, K-group g = L>>4; per ktile two
//   contiguous 16B runs at K = 8g and K = 16+8g -> two ds_load_b128.
// B fragments prepacked so lane L reads 32 contiguous bytes (one v16h).
// C/D (16x16, f32): VGPR i holds row m = i + 8*(L>=16), col n = L&15.
// ---------------------------------------------------------------------------
__device__ inline v16h frag_combine(v8h lo, v8h hi) {
  union { v16h v; v8h h[2]; } u;
  u.h[0] = lo; u.h[1] = hi;
  return u.v;
}

__device__ inline v16h lds_load_a(const _Float16* __restrict__ buf, int lane, int kt) {
  int row = lane & 15;
  int g   = (lane >> 4) & 1;
  const _Float16* p = buf + row * AST + kt * 32;
  v8h lo = *(const v8h*)(p + g * 8);        // K = kt*32 + 8g .. +7
  v8h hi = *(const v8h*)(p + 16 + g * 8);   // K = kt*32 + 16 + 8g .. +7
  return frag_combine(lo, hi);
}

__device__ inline v16h gl_load_b(const _Float16* __restrict__ w, int frag, int lane) {
  return *(const v16h*)(w + (size_t)frag * 512 + lane * 16);
}

__device__ inline float siluf(float x) { return x / (1.0f + __expf(-x)); }
__device__ inline float lreluf(float x) { return x > 0.0f ? x : 0.01f * x; }

__device__ inline v8h pack8(float4 a, float4 b) {
  v8h h;
  h[0] = (_Float16)a.x; h[1] = (_Float16)a.y; h[2] = (_Float16)a.z; h[3] = (_Float16)a.w;
  h[4] = (_Float16)b.x; h[5] = (_Float16)b.y; h[6] = (_Float16)b.z; h[7] = (_Float16)b.w;
  return h;
}

template <int KT, int NT>
__device__ inline void gemm_tile(const _Float16* __restrict__ bufIn,
                                 const _Float16* __restrict__ W,
                                 int lane, v8f* acc) {
#pragma unroll
  for (int nt = 0; nt < NT; ++nt) {
#pragma unroll
    for (int i = 0; i < 8; ++i) acc[nt][i] = 0.0f;
  }
#pragma unroll
  for (int kt = 0; kt < KT; ++kt) {
    v16h a = lds_load_a(bufIn, lane, kt);
#pragma unroll
    for (int nt = 0; nt < NT; ++nt) {
      v16h b = gl_load_b(W, kt * NT + nt, lane);
      acc[nt] = __builtin_amdgcn_wmma_f32_16x16x32_f16(
          false, a, false, b, (short)0, acc[nt], false, false);
    }
  }
}

// ACT: 0 = SiLU, 1 = LeakyReLU(0.01), 2 = identity
template <int NT, int ACT>
__device__ inline void store_act(const v8f* acc, const float* __restrict__ bias,
                                 _Float16* __restrict__ bufOut, int lane) {
  int g  = (lane >> 4) & 1;
  int nl = lane & 15;
#pragma unroll
  for (int nt = 0; nt < NT; ++nt) {
    int n   = nt * 16 + nl;
    float b = bias[n];
#pragma unroll
    for (int i = 0; i < 8; ++i) {
      float x = acc[nt][i] + b;
      if (ACT == 0) x = siluf(x);
      else if (ACT == 1) x = lreluf(x);
      bufOut[(i + 8 * g) * AST + n] = (_Float16)x;
    }
  }
}

// Gather one 64-float node-feature row -> 64 f16 at dstp (16B vector stores).
__device__ inline void gather_row64(const float* __restrict__ rowp,
                                    _Float16* __restrict__ dstp) {
  const float4* s4 = (const float4*)rowp;
#pragma unroll
  for (int q = 0; q < 8; ++q) {
    *(v8h*)(dstp + q * 8) = pack8(s4[2 * q], s4[2 * q + 1]);
  }
}

// ---------------------------------------------------------------------------
// Weight prepack: W[K][N] f32 row-major -> f16 WMMA B fragments.
// frag = kt*NT + nt; 512 halves/fragment; lane owns 16 contiguous halves.
// ---------------------------------------------------------------------------
__global__ void pack_weights_kernel(const float* __restrict__ W,
                                    _Float16* __restrict__ P,
                                    int K, int Nn, int NT) {
  int frag = blockIdx.x;
  int kt = frag / NT, nt = frag % NT;
  int lane = threadIdx.x;         // 0..31
  int n = nt * 16 + (lane & 15);
  int g = (lane >> 4) & 1;
  _Float16* dst = P + (size_t)frag * 512 + lane * 16;
#pragma unroll
  for (int h = 0; h < 16; ++h) {
    int vg = h >> 1, w = h & 1;
    int k = kt * 32 + ((vg & 4) ? 16 : 0) + g * 8 + (vg & 3) * 2 + w;
    float val = (k < K && n < Nn) ? W[(size_t)k * Nn + n] : 0.0f;
    dst[h] = (_Float16)val;
  }
}

// ---------------------------------------------------------------------------
// Workspace zero / misc small kernels
// ---------------------------------------------------------------------------
__global__ void zero_kernel(float* __restrict__ agg, float* __restrict__ deg,
                            int* __restrict__ cnt) {
  int i = blockIdx.x * blockDim.x + threadIdx.x;
  if (i < kN * kD) agg[i] = 0.0f;
  if (i < kN) deg[i] = 0.0f;
  if (i < 8) cnt[i] = 0;
}

__global__ void node_update_kernel(const float* __restrict__ nf,
                                   const float* __restrict__ agg,
                                   const float* __restrict__ deg,
                                   float* __restrict__ n_upd) {
  int i = blockIdx.x * blockDim.x + threadIdx.x;
  if (i >= kN * kD) return;
  int node = i >> 6;
  float dg = fmaxf(deg[node], 1.0f);
  n_upd[i] = agg[i] / dg + nf[i];
}

__global__ void count_kernel(const int* __restrict__ eidx,
                             const int* __restrict__ ax,
                             int* __restrict__ cnt) {
  int e = blockIdx.x * blockDim.x + threadIdx.x;
  if (e >= kE) return;
  int t = ax[eidx[e]] * 2 + ax[eidx[kE + e]];
  atomicAdd(&cnt[t], 1);
}

__global__ void prefix_kernel(const int* __restrict__ cnt, int* __restrict__ cursor) {
  if (blockIdx.x == 0 && threadIdx.x == 0) {
    int s = 0;
    for (int t = 0; t < 4; ++t) { cursor[t] = s; s += cnt[t]; }
  }
}

__global__ void scatter_kernel(const int* __restrict__ eidx,
                               const int* __restrict__ ax,
                               int* __restrict__ cursor,
                               int* __restrict__ list) {
  int e = blockIdx.x * blockDim.x + threadIdx.x;
  if (e >= kE) return;
  int t = ax[eidx[e]] * 2 + ax[eidx[kE + e]];
  int pos = atomicAdd(&cursor[t], 1);
  list[pos] = e;
}

// ---------------------------------------------------------------------------
// Edge phase: edge-update MLP + message MLP + scatter-add aggregation.
// One wave per 16-edge tile; 4 waves / block.
// ---------------------------------------------------------------------------
struct EdgePtrs {
  const float* nf; const float* er; const float* ea; const int* eidx;
  const _Float16 *ew1, *ew2, *ew3, *ew4, *ew5, *ew6, *nw1, *nw2;   // packed f16
  const float *eb1, *eb2, *eb3, *eb4, *eb5, *eb6, *nb1, *nb2;       // f32 biases
  float* e_upd; float* agg; float* deg;
};

__global__ void __launch_bounds__(128) edge_kernel(EdgePtrs p) {
  __shared__ _Float16 smem[4][2][16 * AST];
  int wid = threadIdx.x >> 5, lane = threadIdx.x & 31;
  int tile = blockIdx.x * 4 + wid;   // grid sized exactly: kE/16/4 blocks
  int e0 = tile * 16;
  _Float16* bufA = smem[wid][0];
  _Float16* bufB = smem[wid][1];

  // Build ei = [nf[src] | nf[dst] | radial | angular | 0-pad] : 16 x 160 f16.
  // Lane = (row = lane&15, part = lane>>4): part 0 -> src half, 1 -> dst half.
  {
    int r = lane & 15, part = lane >> 4;
    int e = e0 + r;
    int node = p.eidx[(size_t)part * kE + e];
    gather_row64(p.nf + (size_t)node * kD, bufA + r * AST + part * 64);
    if (part == 0) {
      _Float16* dd = bufA + r * AST + 128;
      float4 r0 = *(const float4*)(p.er + (size_t)e * kR);       // 32B aligned
      float4 r1 = *(const float4*)(p.er + (size_t)e * kR + 4);
      *(v8h*)dd = pack8(r0, r1);
#pragma unroll
      for (int j = 0; j < kA; ++j) dd[8 + j] = (_Float16)p.ea[(size_t)e * kA + j];
#pragma unroll
      for (int j = kC; j < 32; ++j) dd[j] = (_Float16)0.0f;      // pad 145..159
    }
  }
  WAIT_DS();

  v8f acc[8];
  gemm_tile<5, 8>(bufA, p.ew1, lane, acc); store_act<8, 0>(acc, p.eb1, bufB, lane); WAIT_DS();
  gemm_tile<4, 8>(bufB, p.ew2, lane, acc); store_act<8, 0>(acc, p.eb2, bufA, lane); WAIT_DS();
  gemm_tile<4, 4>(bufA, p.ew3, lane, acc); store_act<4, 0>(acc, p.eb3, bufB, lane); WAIT_DS();
  gemm_tile<2, 8>(bufB, p.ew4, lane, acc); store_act<8, 1>(acc, p.eb4, bufA, lane); WAIT_DS();
  gemm_tile<4, 8>(bufA, p.ew5, lane, acc); store_act<8, 0>(acc, p.eb5, bufB, lane); WAIT_DS();
  gemm_tile<4, 2>(bufB, p.ew6, lane, acc);   // N padded 17 -> 32

  // e_upd = acc + eb6 + ef (residual); f32 to ws, f16 into msg cols 64..80.
  {
    int g = (lane >> 4) & 1, nl = lane & 15;
#pragma unroll
    for (int nt = 0; nt < 2; ++nt) {
      int n = nt * 16 + nl;
      if (n < kC) {
#pragma unroll
        for (int i = 0; i < 8; ++i) {
          int m = i + 8 * g;
          int e = e0 + m;
          float efv = (n < kR) ? p.er[(size_t)e * kR + n]
                               : p.ea[(size_t)e * kA + (n - kR)];
          float x = acc[nt][i] + p.eb6[n] + efv;
          p.e_upd[(size_t)e * kC + n] = x;
          bufA[m * AST + 64 + n] = (_Float16)x;
        }
      }
    }
  }
  // msg input cols 0..63 = nf[dst] (float4 gather); zero pad cols 81..95.
  {
    int r = lane >> 1, hf = lane & 1;       // 2 lanes per row, 32 cols each
    int e = e0 + r;
    int d = p.eidx[kE + e];
    const float4* s4 = (const float4*)(p.nf + (size_t)d * kD + hf * 32);
    _Float16* dstp = bufA + r * AST + hf * 32;
#pragma unroll
    for (int q = 0; q < 4; ++q) {
      *(v8h*)(dstp + q * 8) = pack8(s4[2 * q], s4[2 * q + 1]);
    }
    if (lane < 16) {
      _Float16* z = bufA + lane * AST;
#pragma unroll
      for (int j = 81; j < 96; ++j) z[j] = (_Float16)0.0f;
    }
  }
  WAIT_DS();

  // message MLP: 81(pad 96) -> 128 (SiLU) -> 64, then scatter-add into agg.
  gemm_tile<3, 8>(bufA, p.nw1, lane, acc); store_act<8, 0>(acc, p.nb1, bufB, lane); WAIT_DS();
  gemm_tile<4, 4>(bufB, p.nw2, lane, acc);
  {
    int g = (lane >> 4) & 1, nl = lane & 15;
#pragma unroll
    for (int i = 0; i < 8; ++i) {
      int m = i + 8 * g;
      int e = e0 + m;
      int d = p.eidx[kE + e];
      float* aggrow = p.agg + (size_t)d * kD;
#pragma unroll
      for (int nt = 0; nt < 4; ++nt) {
        int n = nt * 16 + nl;
        atomicAdd(aggrow + n, acc[nt][i] + p.nb2[n]);
      }
    }
    if (lane < 16) {
      atomicAdd(&p.deg[p.eidx[kE + e0 + lane]], 1.0f);
    }
  }
}

// ---------------------------------------------------------------------------
// Head phase: per-pair-type extraction MLP over type-bucketed edges.
// ---------------------------------------------------------------------------
struct HeadPtrs {
  const float* n_upd; const float* e_upd; const int* eidx; const int* ax;
  const int* list;
  const _Float16 *w1, *w2, *w3, *w4, *w5;   // packed, 4 heads contiguous
  const float *b1, *b2, *b3, *b4, *b5;      // f32 [4][...]
  float* out;
};

__global__ void __launch_bounds__(128) head_kernel(HeadPtrs p) {
  __shared__ _Float16 smem[4][2][16 * AST];
  __shared__ int s_eid[4][16];
  __shared__ int s_tt[4][16];
  int wid = threadIdx.x >> 5, lane = threadIdx.x & 31;
  int tile = blockIdx.x * 4 + wid;
  int base = tile * 16;
  _Float16* bufA = smem[wid][0];
  _Float16* bufB = smem[wid][1];

  if (lane < 16) {
    int e = p.list[base + lane];
    s_eid[wid][lane] = e;
    s_tt[wid][lane] = p.ax[p.eidx[e]] * 2 + p.ax[p.eidx[kE + e]];
  }
  WAIT_DS();

  unsigned present = 0;
  for (int r = 0; r < 16; ++r) present |= 1u << s_tt[wid][r];

  for (int t = 0; t < 4; ++t) {
    if (!(present & (1u << t))) continue;

    // head_in = [n_upd[src] | n_upd[dst] | e_upd | 0-pad] : 16 x 160 f16
    {
      int r = lane & 15, part = lane >> 4;
      int e = s_eid[wid][r];
      int node = p.eidx[(size_t)part * kE + e];
      gather_row64(p.n_upd + (size_t)node * kD, bufA + r * AST + part * 64);
      if (part == 0) {
        _Float16* dd = bufA + r * AST + 128;
        const float* eu = p.e_upd + (size_t)e * kC;   // 17 floats, 4B aligned
#pragma unroll
        for (int j = 0; j < kC; ++j) dd[j] = (_Float16)eu[j];
#pragma unroll
        for (int j = kC; j < 32; ++j) dd[j] = (_Float16)0.0f;
      }
    }
    WAIT_DS();

    v8f acc[8];
    const _Float16* w1 = p.w1 + (size_t)t * 40 * 512;
    const _Float16* w2 = p.w2 + (size_t)t * 32 * 512;
    const _Float16* w3 = p.w3 + (size_t)t * 32 * 512;
    const _Float16* w4 = p.w4 + (size_t)t * 32 * 512;
    const _Float16* w5 = p.w5 + (size_t)t * 24 * 512;
    gemm_tile<5, 8>(bufA, w1, lane, acc); store_act<8, 0>(acc, p.b1 + t * kH, bufB, lane); WAIT_DS();
    gemm_tile<4, 8>(bufB, w2, lane, acc); store_act<8, 0>(acc, p.b2 + t * kH, bufA, lane); WAIT_DS();
    gemm_tile<4, 8>(bufA, w3, lane, acc); store_act<8, 0>(acc, p.b3 + t * kH, bufB, lane); WAIT_DS();
    gemm_tile<4, 8>(bufB, w4, lane, acc); store_act<8, 1>(acc, p.b4 + t * kH, bufA, lane); WAIT_DS();
    gemm_tile<4, 6>(bufA, w5, lane, acc);   // N padded 81 -> 96

    {
      int g = (lane >> 4) & 1, nl = lane & 15;
#pragma unroll
      for (int i = 0; i < 8; ++i) {
        int m = i + 8 * g;
        if (s_tt[wid][m] != t) continue;
        float* orow = p.out + (size_t)s_eid[wid][m] * kNO2;
#pragma unroll
        for (int nt = 0; nt < 6; ++nt) {
          int n = nt * 16 + nl;
          if (n < kNO2) orow[n] = acc[nt][i] + p.b5[t * kNO2 + n];
        }
      }
    }
  }
}

// ---------------------------------------------------------------------------
// Host-side launch
// ---------------------------------------------------------------------------
extern "C" void kernel_launch(void* const* d_in, const int* in_sizes, int n_in,
                              void* d_out, int out_size, void* d_ws, size_t ws_size,
                              hipStream_t stream) {
  (void)in_sizes; (void)n_in; (void)out_size; (void)ws_size;

  // ---- inputs (setup_inputs dict order, nested dicts flattened in order) ----
  const float* nf  = (const float*)d_in[0];
  const float* er  = (const float*)d_in[1];
  const float* ea  = (const float*)d_in[2];
  const int*   eix = (const int*)d_in[3];
  const int*   ax  = (const int*)d_in[4];
  const float* nw1 = (const float*)d_in[5];   const float* nb1 = (const float*)d_in[6];
  const float* nw2 = (const float*)d_in[7];   const float* nb2 = (const float*)d_in[8];
  const float* ew1 = (const float*)d_in[9];   const float* eb1 = (const float*)d_in[10];
  const float* ew2 = (const float*)d_in[11];  const float* eb2 = (const float*)d_in[12];
  const float* ew3 = (const float*)d_in[13];  const float* eb3 = (const float*)d_in[14];
  const float* ew4 = (const float*)d_in[15];  const float* eb4 = (const float*)d_in[16];
  const float* ew5 = (const float*)d_in[17];  const float* eb5 = (const float*)d_in[18];
  const float* ew6 = (const float*)d_in[19];  const float* eb6 = (const float*)d_in[20];
  const float* hw1 = (const float*)d_in[21];  const float* hb1 = (const float*)d_in[22];
  const float* hw2 = (const float*)d_in[23];  const float* hb2 = (const float*)d_in[24];
  const float* hw3 = (const float*)d_in[25];  const float* hb3 = (const float*)d_in[26];
  const float* hw4 = (const float*)d_in[27];  const float* hb4 = (const float*)d_in[28];
  const float* hw5 = (const float*)d_in[29];  const float* hb5 = (const float*)d_in[30];
  float* out = (float*)d_out;

  // ---- workspace layout (bump allocator, all offsets 32B aligned) ----
  char* ws = (char*)d_ws;
  size_t off = 0;
  auto bump = [&](size_t bytes) { void* p = ws + off; off = (off + bytes + 31) & ~(size_t)31; return p; };

  auto frag_bytes = [](int KT, int NT) { return (size_t)KT * NT * 512 * sizeof(_Float16); };
  _Float16* ew1p = (_Float16*)bump(frag_bytes(5, 8));
  _Float16* ew2p = (_Float16*)bump(frag_bytes(4, 8));
  _Float16* ew3p = (_Float16*)bump(frag_bytes(4, 4));
  _Float16* ew4p = (_Float16*)bump(frag_bytes(2, 8));
  _Float16* ew5p = (_Float16*)bump(frag_bytes(4, 8));
  _Float16* ew6p = (_Float16*)bump(frag_bytes(4, 2));
  _Float16* nw1p = (_Float16*)bump(frag_bytes(3, 8));
  _Float16* nw2p = (_Float16*)bump(frag_bytes(4, 4));
  _Float16* hw1p = (_Float16*)bump(4 * frag_bytes(5, 8));
  _Float16* hw2p = (_Float16*)bump(4 * frag_bytes(4, 8));
  _Float16* hw3p = (_Float16*)bump(4 * frag_bytes(4, 8));
  _Float16* hw4p = (_Float16*)bump(4 * frag_bytes(4, 8));
  _Float16* hw5p = (_Float16*)bump(4 * frag_bytes(4, 6));
  float* e_upd  = (float*)bump((size_t)kE * kC * sizeof(float));
  float* agg    = (float*)bump((size_t)kN * kD * sizeof(float));
  float* deg    = (float*)bump((size_t)kN * sizeof(float));
  float* n_upd  = (float*)bump((size_t)kN * kD * sizeof(float));
  int*   cnt    = (int*)bump(8 * sizeof(int));      // cnt[0..3], cursor[4..7]
  int*   list   = (int*)bump((size_t)kE * sizeof(int));
  int*   cursor = cnt + 4;

  // ---- 1) zero accumulators / counters ----
  zero_kernel<<<(kN * kD + 255) / 256, 256, 0, stream>>>(agg, deg, cnt);

  // ---- 2) prepack weights to f16 WMMA B-fragments ----
  auto pack = [&](const float* W, _Float16* P, int K, int Nn, int KT, int NT) {
    pack_weights_kernel<<<KT * NT, 32, 0, stream>>>(W, P, K, Nn, NT);
  };
  pack(ew1, ew1p, kHI, kH, 5, 8);
  pack(ew2, ew2p, kH,  kH, 4, 8);
  pack(ew3, ew3p, kH,  64, 4, 4);
  pack(ew4, ew4p, 64,  kH, 2, 8);
  pack(ew5, ew5p, kH,  kH, 4, 8);
  pack(ew6, ew6p, kH,  kC, 4, 2);
  pack(nw1, nw1p, kD + kC, kH, 3, 8);
  pack(nw2, nw2p, kH,  kD, 4, 4);
  for (int h = 0; h < 4; ++h) {
    pack(hw1 + (size_t)h * kHI * kH, hw1p + (size_t)h * 40 * 512, kHI, kH,   5, 8);
    pack(hw2 + (size_t)h * kH * kH,  hw2p + (size_t)h * 32 * 512, kH,  kH,   4, 8);
    pack(hw3 + (size_t)h * kH * kH,  hw3p + (size_t)h * 32 * 512, kH,  kH,   4, 8);
    pack(hw4 + (size_t)h * kH * kH,  hw4p + (size_t)h * 32 * 512, kH,  kH,   4, 8);
    pack(hw5 + (size_t)h * kH * kNO2, hw5p + (size_t)h * 24 * 512, kH, kNO2, 4, 6);
  }

  // ---- 3) edge phase (edge MLP + message MLP + scatter-add) ----
  EdgePtrs ep;
  ep.nf = nf; ep.er = er; ep.ea = ea; ep.eidx = eix;
  ep.ew1 = ew1p; ep.ew2 = ew2p; ep.ew3 = ew3p; ep.ew4 = ew4p; ep.ew5 = ew5p; ep.ew6 = ew6p;
  ep.nw1 = nw1p; ep.nw2 = nw2p;
  ep.eb1 = eb1; ep.eb2 = eb2; ep.eb3 = eb3; ep.eb4 = eb4; ep.eb5 = eb5; ep.eb6 = eb6;
  ep.nb1 = nb1; ep.nb2 = nb2;
  ep.e_upd = e_upd; ep.agg = agg; ep.deg = deg;
  edge_kernel<<<kE / 16 / 4, 128, 0, stream>>>(ep);

  // ---- 4) node update ----
  node_update_kernel<<<(kN * kD + 255) / 256, 256, 0, stream>>>(nf, agg, deg, n_upd);

  // ---- 5) bucket edges by pair type ----
  count_kernel<<<(kE + 255) / 256, 256, 0, stream>>>(eix, ax, cnt);
  prefix_kernel<<<1, 32, 0, stream>>>(cnt, cursor);
  scatter_kernel<<<(kE + 255) / 256, 256, 0, stream>>>(eix, ax, cursor, list);

  // ---- 6) head phase ----
  HeadPtrs hp;
  hp.n_upd = n_upd; hp.e_upd = e_upd; hp.eidx = eix; hp.ax = ax; hp.list = list;
  hp.w1 = hw1p; hp.w2 = hw2p; hp.w3 = hw3p; hp.w4 = hw4p; hp.w5 = hw5p;
  hp.b1 = hb1; hp.b2 = hb2; hp.b3 = hb3; hp.b4 = hb4; hp.b5 = hb5;
  hp.out = out;
  head_kernel<<<kE / 16 / 4, 128, 0, stream>>>(hp);
}
